// ParallelBlock_24842090840769
// MI455X (gfx1250) — matrix operand order repaired and checked by reading, not compiled, not run on hardware
//
#include <hip/hip_runtime.h>
#include <hip/hip_bf16.h>

typedef _Float16 h16;
typedef __attribute__((ext_vector_type(16))) _Float16 v16h;
typedef __attribute__((ext_vector_type(8)))  _Float16 v8h;
typedef __attribute__((ext_vector_type(8)))  float    v8f;

#define DIM       1024
#define HEADS     16
#define HEAD_DIM  64
#define SEQ       2048
#define BATCH     2
#define ROWS      (BATCH * SEQ)     // 4096
#define MLP_HID   716
#define MLP_PAD   736               // 23*32
#define FUSED_N   3788
#define FUSED_PAD 3840              // 15*256

// ---------------------------------------------------------------- WMMA helpers
__device__ __forceinline__ v8f wmma_f16(v16h a, v16h b, v8f c) {
  return __builtin_amdgcn_wmma_f32_16x16x32_f16(
      /*neg_a=*/false, a, /*neg_b=*/false, b,
      /*c_mod=*/(short)0, c, /*reuse_a=*/false, /*reuse_b=*/false);
}

// A fragment (16x32 f16): p = rowptr + kbase + 8*g ; halves [0..7]=K+0..7,
// halves [8..15]=K+16..23 (per ISA 16-bit A layout).
__device__ __forceinline__ v16h load_a_frag(const h16* __restrict__ p) {
  union { v16h v; v8h h[2]; } u;
  u.h[0] = *(const v8h*)(p);
  u.h[1] = *(const v8h*)(p + 16);
  return u.v;
}
// B fragment (32x16 f16) from K-major storage: p = colptr + kbase + 8*g ;
// halves [0..7]=K+0..7, halves [8..15]=K+16..23 (same order as A).
__device__ __forceinline__ v16h load_b_frag(const h16* __restrict__ p) {
  union { v16h v; v8h h[2]; } u;
  u.h[0] = *(const v8h*)(p);
  u.h[1] = *(const v8h*)(p + 16);
  return u.v;
}

// ---------------------------------------------------------------- weight prep
// W[Ksrc][Nsrc] f32 row-major -> Wt[Npad][Kpad] f16 (K-major), zero padded.
__global__ void convw_kernel(const float* __restrict__ W, int Ksrc, int Nsrc,
                             h16* __restrict__ Wt, int Kpad, int Npad) {
  size_t i = (size_t)blockIdx.x * blockDim.x + threadIdx.x;
  size_t total = (size_t)Kpad * Npad;
  if (i >= total) return;
  int k = (int)(i % Kpad);
  int n = (int)(i / Kpad);
  float v = (k < Ksrc && n < Nsrc) ? W[(size_t)k * Nsrc + n] : 0.0f;
  Wt[i] = (h16)v;
}

// ---------------------------------------------------------------- input LN
__global__ __launch_bounds__(256) void lnx_kernel(const float* __restrict__ x,
                                                  const float* __restrict__ g,
                                                  const float* __restrict__ bb,
                                                  h16* __restrict__ out) {
  int row = blockIdx.x;
  const float* xr = x + (size_t)row * DIM;
  float v[4], s = 0.f, ss = 0.f;
#pragma unroll
  for (int i = 0; i < 4; i++) {
    v[i] = xr[threadIdx.x + i * 256];
    s += v[i]; ss += v[i] * v[i];
  }
#pragma unroll
  for (int m = 16; m >= 1; m >>= 1) {
    s  += __shfl_xor(s,  m);
    ss += __shfl_xor(ss, m);
  }
  __shared__ float sh[16];
  int lane = threadIdx.x & 31, wave = threadIdx.x >> 5;
  if (lane == 0) { sh[wave] = s; sh[8 + wave] = ss; }
  __syncthreads();
  float ts = 0.f, tss = 0.f;
#pragma unroll
  for (int w = 0; w < 8; w++) { ts += sh[w]; tss += sh[8 + w]; }
  float mu  = ts  * (1.f / 1024.f);
  float var = tss * (1.f / 1024.f) - mu * mu;
  float rs  = rsqrtf(var + 1e-6f);
#pragma unroll
  for (int i = 0; i < 4; i++) {
    int c = threadIdx.x + i * 256;
    out[(size_t)row * DIM + c] = (h16)((v[i] - mu) * rs * g[c] + bb[c]);
  }
}

// ---------------------------------------------------------------- WMMA GEMM
// C[M,N] = A[M,K](f16,row-major) x Bt[N,K](f16,K-major) + bias
// mode 0: store f16 to C16 ; mode 1: store f32 = acc + bias + resid
// Block tile 64(M) x 256(N): 8 waves (2 M x 4 N), wave tile 32x64.
// K-loop is software-pipelined: next fragments are issued before the 8 WMMAs
// on the current fragments, hiding load latency behind matrix math.
__global__ __launch_bounds__(256) void gemm_kernel(
    const h16* __restrict__ A, int lda,
    const h16* __restrict__ Bt, int ldb, int K,
    const float* __restrict__ bias, int nbias,
    h16* __restrict__ C16, float* __restrict__ C32,
    const float* __restrict__ resid, int ldc, int mode) {
  const int lane = threadIdx.x & 31;
  const int wave = threadIdx.x >> 5;
  const int g    = lane >> 4;
  const int n16  = lane & 15;
  const int mw   = wave & 1;           // 2 M-waves
  const int nw   = wave >> 1;          // 4 N-waves
  const int mbase = blockIdx.y * 64 + mw * 32;
  const int nbase = blockIdx.x * 256 + nw * 64;

  v8f acc[2][4] = {};
  const h16* arow0 = A + (size_t)(mbase + n16) * lda + 8 * g;
  const h16* arow1 = arow0 + (size_t)16 * lda;
  const h16* brow[4];
#pragma unroll
  for (int j = 0; j < 4; j++)
    brow[j] = Bt + (size_t)(nbase + j * 16 + n16) * ldb + 8 * g;

  v16h a0 = load_a_frag(arow0);
  v16h a1 = load_a_frag(arow1);
  v16h b0 = load_b_frag(brow[0]);
  v16h b1 = load_b_frag(brow[1]);
  v16h b2 = load_b_frag(brow[2]);
  v16h b3 = load_b_frag(brow[3]);

  for (int k = 32; k < K; k += 32) {
    if (k + 128 < K) __builtin_prefetch(brow[0] + k + 128, 0, 3);
    // issue next-k fragment loads first (overlap with WMMAs below)
    v16h na0 = load_a_frag(arow0 + k);
    v16h na1 = load_a_frag(arow1 + k);
    v16h nb0 = load_b_frag(brow[0] + k);
    v16h nb1 = load_b_frag(brow[1] + k);
    v16h nb2 = load_b_frag(brow[2] + k);
    v16h nb3 = load_b_frag(brow[3] + k);

    acc[0][0] = wmma_f16(a0, b0, acc[0][0]);
    acc[1][0] = wmma_f16(a1, b0, acc[1][0]);
    acc[0][1] = wmma_f16(a0, b1, acc[0][1]);
    acc[1][1] = wmma_f16(a1, b1, acc[1][1]);
    acc[0][2] = wmma_f16(a0, b2, acc[0][2]);
    acc[1][2] = wmma_f16(a1, b2, acc[1][2]);
    acc[0][3] = wmma_f16(a0, b3, acc[0][3]);
    acc[1][3] = wmma_f16(a1, b3, acc[1][3]);

    a0 = na0; a1 = na1; b0 = nb0; b1 = nb1; b2 = nb2; b3 = nb3;
  }
  // drain last fragments
  acc[0][0] = wmma_f16(a0, b0, acc[0][0]);
  acc[1][0] = wmma_f16(a1, b0, acc[1][0]);
  acc[0][1] = wmma_f16(a0, b1, acc[0][1]);
  acc[1][1] = wmma_f16(a1, b1, acc[1][1]);
  acc[0][2] = wmma_f16(a0, b2, acc[0][2]);
  acc[1][2] = wmma_f16(a1, b2, acc[1][2]);
  acc[0][3] = wmma_f16(a0, b3, acc[0][3]);
  acc[1][3] = wmma_f16(a1, b3, acc[1][3]);

  int   ng[4];
  float bv[4];
#pragma unroll
  for (int j = 0; j < 4; j++) {
    ng[j] = nbase + j * 16 + n16;
    bv[j] = (ng[j] < nbias) ? bias[ng[j]] : 0.0f;
  }
#pragma unroll
  for (int i = 0; i < 2; i++) {
#pragma unroll
    for (int r = 0; r < 8; r++) {
      int m = mbase + i * 16 + r + 8 * g;
      size_t off = (size_t)m * ldc;
      if (mode == 0) {
#pragma unroll
        for (int j = 0; j < 4; j++)
          C16[off + ng[j]] = (h16)(acc[i][j][r] + bv[j]);
      } else {
#pragma unroll
        for (int j = 0; j < 4; j++)
          C32[off + ng[j]] = acc[i][j][r] + bv[j] + resid[off + ng[j]];
      }
    }
  }
}

// ---------------------------------------------------------------- QK-LN pack
// fused[row][FUSED_PAD] f16 -> Qp/Kp [bh][n][64] f16 (QK-layernormed),
// Vt [bh][64][SEQ] f16 (transposed for PV B-operand). One wave per (b,h,n).
__global__ __launch_bounds__(256) void qkpack_kernel(
    const h16* __restrict__ fused,
    const float* __restrict__ qg, const float* __restrict__ qb,
    const float* __restrict__ kg, const float* __restrict__ kb_,
    h16* __restrict__ Qp, h16* __restrict__ Kp, h16* __restrict__ Vt) {
  int lane = threadIdx.x & 31, wave = threadIdx.x >> 5;
  int idx = blockIdx.x * 8 + wave;     // (b,h,n)
  int b  = idx >> 15;
  int hn = idx & 32767;
  int h  = hn >> 11;
  int n  = hn & 2047;
  const h16* frow = fused + (size_t)(b * SEQ + n) * FUSED_PAD;
  int d0 = lane * 2;
  size_t qko = ((size_t)(b * HEADS + h) * SEQ + n) * HEAD_DIM + d0;

#pragma unroll
  for (int which = 0; which < 2; which++) {   // 0 = q, 1 = k
    int cbase = which * DIM + h * HEAD_DIM;
    float x0 = (float)frow[cbase + d0];
    float x1 = (float)frow[cbase + d0 + 1];
    float s = x0 + x1, ss = x0 * x0 + x1 * x1;
#pragma unroll
    for (int m = 16; m >= 1; m >>= 1) {
      s  += __shfl_xor(s,  m);
      ss += __shfl_xor(ss, m);
    }
    float mu  = s  * (1.f / 64.f);
    float var = ss * (1.f / 64.f) - mu * mu;
    float rs  = rsqrtf(var + 1e-6f);
    const float* gg = which ? kg : qg;
    const float* bb = which ? kb_ : qb;
    h16* dst = which ? Kp : Qp;
    dst[qko]     = (h16)((x0 - mu) * rs * gg[d0]     + bb[d0]);
    dst[qko + 1] = (h16)((x1 - mu) * rs * gg[d0 + 1] + bb[d0 + 1]);
  }
  // v: plain convert, transposed [bh][d][n]
  {
    float v0 = (float)frow[2 * DIM + h * HEAD_DIM + d0];
    float v1 = (float)frow[2 * DIM + h * HEAD_DIM + d0 + 1];
    size_t base = (size_t)(b * HEADS + h) * HEAD_DIM;
    Vt[(base + d0)     * SEQ + n] = (h16)v0;
    Vt[(base + d0 + 1) * SEQ + n] = (h16)v1;
  }
}

// ---------------------------------------------------------------- attention
// One wave = one 16-row q-tile of one (b,h). Flash-style online softmax over
// 32-key steps; P goes through LDS to convert D-layout -> A-layout.
__global__ __launch_bounds__(256) void attn_kernel(
    const h16* __restrict__ Qp, const h16* __restrict__ Kp,
    const h16* __restrict__ Vt, h16* __restrict__ Ah) {
  __shared__ h16 pbuf[8][16][32];      // 8 KB, wave-private slices
  const int lane = threadIdx.x & 31, wave = threadIdx.x >> 5;
  const int g = lane >> 4, n16 = lane & 15;
  int tile  = blockIdx.x * 8 + wave;   // 4096 q-tiles total
  int bh    = tile >> 7;               // SEQ/16 = 128 tiles per (b,h)
  int qbase = (tile & 127) * 16;
  const h16* Qb = Qp + (size_t)bh * SEQ * HEAD_DIM;
  const h16* Kb = Kp + (size_t)bh * SEQ * HEAD_DIM;
  const h16* Vb = Vt + (size_t)bh * HEAD_DIM * SEQ;

  v16h qa0 = load_a_frag(Qb + (size_t)(qbase + n16) * HEAD_DIM + 0  + 8 * g);
  v16h qa1 = load_a_frag(Qb + (size_t)(qbase + n16) * HEAD_DIM + 32 + 8 * g);

  v8f acc[4] = {};
  float mrow[8], lrow[8];
#pragma unroll
  for (int r = 0; r < 8; r++) { mrow[r] = -1e30f; lrow[r] = 0.f; }

  for (int kb = 0; kb < SEQ; kb += 32) {
    const h16* kr0 = Kb + (size_t)(kb + n16)      * HEAD_DIM + 8 * g;
    const h16* kr1 = Kb + (size_t)(kb + 16 + n16) * HEAD_DIM + 8 * g;
    if (kb + 32 < SEQ) {
      __builtin_prefetch(kr0 + 32 * HEAD_DIM, 0, 3);
      __builtin_prefetch(kr1 + 32 * HEAD_DIM, 0, 3);
    }
    // S = Q K^T for two 16-key tiles (B operand: Kp row = key, contiguous dh)
    v8f s0 = {}, s1 = {};
    s0 = wmma_f16(qa0, load_b_frag(kr0),      s0);
    s0 = wmma_f16(qa1, load_b_frag(kr0 + 32), s0);
    s1 = wmma_f16(qa0, load_b_frag(kr1),      s1);
    s1 = wmma_f16(qa1, load_b_frag(kr1 + 32), s1);

    const float sc = 0.125f;             // HEAD_DIM^-0.5
#pragma unroll
    for (int r = 0; r < 8; r++) {
      float a0 = s0[r] * sc, a1 = s1[r] * sc;
      float rm = fmaxf(a0, a1);
      rm = fmaxf(rm, __shfl_xor(rm, 1));
      rm = fmaxf(rm, __shfl_xor(rm, 2));
      rm = fmaxf(rm, __shfl_xor(rm, 4));
      rm = fmaxf(rm, __shfl_xor(rm, 8));
      float mn = fmaxf(mrow[r], rm);
      float p0 = __expf(a0 - mn);
      float p1 = __expf(a1 - mn);
      float ps = p0 + p1;
      ps += __shfl_xor(ps, 1);
      ps += __shfl_xor(ps, 2);
      ps += __shfl_xor(ps, 4);
      ps += __shfl_xor(ps, 8);
      float scl = __expf(mrow[r] - mn);
      lrow[r] = lrow[r] * scl + ps;
      mrow[r] = mn;
#pragma unroll
      for (int j = 0; j < 4; j++) acc[j][r] *= scl;
      // D-layout (m = r+8g, n = n16) -> plain [m][key] in LDS
      pbuf[wave][r + 8 * g][n16]      = (h16)p0;
      pbuf[wave][r + 8 * g][16 + n16] = (h16)p1;
    }
    // Re-read P as A fragment (m = n16, two contiguous 8-half runs)
    v16h pa;
    {
      const h16* prow = &pbuf[wave][n16][8 * g];
      union { v16h v; v8h h[2]; } u;
      u.h[0] = *(const v8h*)(prow);
      u.h[1] = *(const v8h*)(prow + 16);
      pa = u.v;
    }
    // PV: B operand from Vt[dh][key] — two 16-byte runs per lane
#pragma unroll
    for (int j = 0; j < 4; j++) {
      v16h bv = load_b_frag(Vb + (size_t)(j * 16 + n16) * SEQ + kb + 8 * g);
      acc[j] = wmma_f16(pa, bv, acc[j]);
    }
  }

  int b = bh >> 4, h = bh & 15;
#pragma unroll
  for (int r = 0; r < 8; r++) {
    float inv = 1.0f / lrow[r];
    size_t row = (size_t)(b * SEQ + qbase + r + 8 * g);
#pragma unroll
    for (int j = 0; j < 4; j++)
      Ah[row * DIM + h * HEAD_DIM + j * 16 + n16] = (h16)(acc[j][r] * inv);
  }
}

// ---------------------------------------------------------------- GELU
__global__ void gelu_kernel(const h16* __restrict__ fused, h16* __restrict__ out) {
  int i = blockIdx.x * blockDim.x + threadIdx.x;
  int total = ROWS * MLP_PAD;
  if (i >= total) return;
  int row = i / MLP_PAD, col = i % MLP_PAD;
  float y = 0.0f;
  if (col < MLP_HID) {
    float x = (float)fused[(size_t)row * FUSED_PAD + 3 * DIM + col];
    float t = tanhf(0.7978845608f * (x + 0.044715f * x * x * x));
    y = 0.5f * x * (1.0f + t);
  }
  out[(size_t)row * MLP_PAD + col] = (h16)y;
}

// ---------------------------------------------------------------- launch
extern "C" void kernel_launch(void* const* d_in, const int* in_sizes, int n_in,
                              void* d_out, int out_size, void* d_ws, size_t ws_size,
                              hipStream_t stream) {
  const float* x   = (const float*)d_in[0];
  const float* ng  = (const float*)d_in[1];
  const float* nb  = (const float*)d_in[2];
  const float* W1  = (const float*)d_in[3];
  const float* b1  = (const float*)d_in[4];
  const float* qng = (const float*)d_in[5];
  const float* qnb = (const float*)d_in[6];
  const float* kng = (const float*)d_in[7];
  const float* knb = (const float*)d_in[8];
  const float* Wao = (const float*)d_in[9];
  const float* bao = (const float*)d_in[10];
  const float* Wmo = (const float*)d_in[11];
  const float* bmo = (const float*)d_in[12];
  const float* W2  = (const float*)d_in[13];
  const float* b2  = (const float*)d_in[14];
  float* out = (float*)d_out;

  char* ws = (char*)d_ws;
  size_t off = 0;
  auto alloc = [&](size_t bytes) -> char* {
    char* p = ws + off;
    off += (bytes + 255) & ~(size_t)255;
    return p;
  };
  h16* normx = (h16*)alloc((size_t)ROWS * DIM * 2);
  h16* W1t   = (h16*)alloc((size_t)FUSED_PAD * DIM * 2);
  h16* Waot  = (h16*)alloc((size_t)DIM * DIM * 2);
  h16* Wmot  = (h16*)alloc((size_t)DIM * MLP_PAD * 2);
  h16* W2t   = (h16*)alloc((size_t)DIM * 2 * DIM * 2);
  h16* fused = (h16*)alloc((size_t)ROWS * FUSED_PAD * 2);
  h16* Qp    = (h16*)alloc((size_t)BATCH * HEADS * SEQ * HEAD_DIM * 2);
  h16* Kp    = (h16*)alloc((size_t)BATCH * HEADS * SEQ * HEAD_DIM * 2);
  h16* Vt    = (h16*)alloc((size_t)BATCH * HEADS * SEQ * HEAD_DIM * 2);
  h16* Ah    = (h16*)alloc((size_t)ROWS * DIM * 2);
  h16* Gh    = (h16*)alloc((size_t)ROWS * MLP_PAD * 2);
  h16* Ch    = (h16*)alloc((size_t)ROWS * 2 * DIM * 2);

  // weights -> f16 [N][K]
  convw_kernel<<<15360, 256, 0, stream>>>(W1, DIM, FUSED_N, W1t, DIM, FUSED_PAD);
  convw_kernel<<<4096,  256, 0, stream>>>(Wao, DIM, DIM, Waot, DIM, DIM);
  convw_kernel<<<2944,  256, 0, stream>>>(Wmo, MLP_HID, DIM, Wmot, MLP_PAD, DIM);
  convw_kernel<<<8192,  256, 0, stream>>>(W2, 2 * DIM, DIM, W2t, 2 * DIM, DIM);

  // layernorm(x) -> f16
  lnx_kernel<<<ROWS, 256, 0, stream>>>(x, ng, nb, normx);

  // fused = normx @ W1 + b1   [4096 x 3840]
  dim3 g1(FUSED_PAD / 256, ROWS / 64);
  gemm_kernel<<<g1, 256, 0, stream>>>(normx, DIM, W1t, DIM, DIM,
                                      b1, FUSED_N, fused, nullptr, nullptr,
                                      FUSED_PAD, 0);

  // per-head QK layernorm + pack q/k/v
  qkpack_kernel<<<(BATCH * HEADS * SEQ) / 8, 256, 0, stream>>>(
      fused, qng, qnb, kng, knb, Qp, Kp, Vt);

  // attention -> Ah [4096 x 1024] f16
  attn_kernel<<<(BATCH * HEADS * (SEQ / 16)) / 8, 256, 0, stream>>>(Qp, Kp, Vt, Ah);

  // gelu(mlp_hidden) -> Gh [4096 x 736] f16
  gelu_kernel<<<(ROWS * MLP_PAD) / 256, 256, 0, stream>>>(fused, Gh);

  // combined[:, :1024]  = Ah @ Wao + bao
  // combined[:, 1024:]  = Gh @ Wmo + bmo
  dim3 g2(DIM / 256, ROWS / 64);
  gemm_kernel<<<g2, 256, 0, stream>>>(Ah, DIM, Waot, DIM, DIM,
                                      bao, DIM, Ch, nullptr, nullptr, 2 * DIM, 0);
  gemm_kernel<<<g2, 256, 0, stream>>>(Gh, MLP_PAD, Wmot, MLP_PAD, MLP_PAD,
                                      bmo, DIM, Ch + DIM, nullptr, nullptr, 2 * DIM, 0);

  // out = x + combined @ W2 + b2   (f32)
  gemm_kernel<<<g2, 256, 0, stream>>>(Ch, 2 * DIM, W2t, 2 * DIM, 2 * DIM,
                                      b2, DIM, nullptr, out, x, DIM, 1);
}
